// NAFLTEOURS_84284438216727
// MI455X (gfx1250) — compile-verified, matrix operand order checked by reading, not run on hardware
//
#include <hip/hip_runtime.h>
#include <math.h>

#define B_   2
#define C_   64
#define H_   96
#define W_   96
#define HID_ 128
#define HH_  384
#define WW_  384
#define Q_   (HH_*WW_)
#define NPIX (H_*W_)

#define PI_F 3.14159265358979323846f

typedef __attribute__((ext_vector_type(16))) _Float16 v16h;
typedef __attribute__((ext_vector_type(8)))  _Float16 v8h;
typedef __attribute__((ext_vector_type(8)))  float    v8f;

// ---------------------------------------------------------------------------
// 1) softmax over last dim (96) of M_right_to_left : rows = B*H*W
// ---------------------------------------------------------------------------
__global__ void k_softmax(const float* __restrict__ M, float* __restrict__ att) {
  const int wave = threadIdx.x >> 5;
  const int lane = threadIdx.x & 31;
  const int row  = blockIdx.x * 4 + wave;
  if (row >= B_ * H_ * W_) return;
  const float* src = M + (size_t)row * W_;
  float v0 = src[lane], v1 = src[lane + 32], v2 = src[lane + 64];
  float mx = fmaxf(v0, fmaxf(v1, v2));
  #pragma unroll
  for (int off = 16; off >= 1; off >>= 1) mx = fmaxf(mx, __shfl_xor(mx, off, 32));
  float e0 = __expf(v0 - mx), e1 = __expf(v1 - mx), e2 = __expf(v2 - mx);
  float s = e0 + e1 + e2;
  #pragma unroll
  for (int off = 16; off >= 1; off >>= 1) s += __shfl_xor(s, off, 32);
  float inv = 1.0f / s;
  float* dst = att + (size_t)row * W_;
  dst[lane] = e0 * inv; dst[lane + 32] = e1 * inv; dst[lane + 64] = e2 * inv;
}

// ---------------------------------------------------------------------------
// 2) fr = projr(feat_right)  (1x1 conv, C->C)
// ---------------------------------------------------------------------------
__global__ void k_proj(const float* __restrict__ fR, const float* __restrict__ wp,
                       const float* __restrict__ bp, float* __restrict__ fr) {
  const int idx = blockIdx.x * blockDim.x + threadIdx.x;
  if (idx >= B_ * C_ * NPIX) return;
  const int b = idx / (C_ * NPIX);
  const int o = (idx / NPIX) % C_;
  const int p = idx % NPIX;
  float acc = bp[o];
  const float* base = fR + (size_t)b * C_ * NPIX + p;
  const float* wrow = wp + (size_t)o * C_;
  for (int c = 0; c < C_; ++c) acc += wrow[c] * base[(size_t)c * NPIX];
  fr[idx] = acc;
}

// ---------------------------------------------------------------------------
// 3) feat = feat_left + att @ fr
//    feat[b,c,y,x] = fL[...] + sum_k att[b,y,x,k] * fr[b,c,y,k]
// ---------------------------------------------------------------------------
__global__ void k_warp(const float* __restrict__ fL, const float* __restrict__ fr,
                       const float* __restrict__ att, float* __restrict__ feat) {
  const int idx = blockIdx.x * blockDim.x + threadIdx.x;
  if (idx >= B_ * C_ * NPIX) return;
  const int b  = idx / (C_ * NPIX);
  const int ch = (idx / NPIX) % C_;
  const int p  = idx % NPIX;
  const int y = p / W_, x = p % W_;
  const float* arow = att + ((((size_t)b * H_ + y) * W_ + x) * W_);
  const float* frow = fr  + ((size_t)b * C_ + ch) * NPIX + (size_t)y * W_;
  float acc = fL[idx];
  for (int k = 0; k < W_; ++k) acc += arow[k] * frow[k];
  feat[idx] = acc;
}

// ---------------------------------------------------------------------------
// 4) coef/freq = conv3x3(feat) (SAME, zero pad), OIHW weights, C=64 -> HID=128
// ---------------------------------------------------------------------------
__global__ void k_conv3(const float* __restrict__ feat,
                        const float* __restrict__ wc, const float* __restrict__ bc,
                        const float* __restrict__ wf, const float* __restrict__ bf,
                        float* __restrict__ coef, float* __restrict__ freq) {
  const int idx = blockIdx.x * blockDim.x + threadIdx.x;
  const int per = B_ * HID_ * NPIX;
  if (idx >= 2 * per) return;
  const int which = idx / per;
  const int rem = idx % per;
  const int b = rem / (HID_ * NPIX);
  const int o = (rem / NPIX) % HID_;
  const int p = rem % NPIX;
  const int y = p / W_, x = p % W_;
  const float* w  = which ? wf : wc;
  const float* bb = which ? bf : bc;
  float acc = bb[o];
  for (int c = 0; c < C_; ++c) {
    const float* fp = feat + ((size_t)b * C_ + c) * NPIX;
    const float* wp = w + ((size_t)o * C_ + c) * 9;
    #pragma unroll
    for (int dy = -1; dy <= 1; ++dy) {
      const int yy = y + dy;
      if (yy < 0 || yy >= H_) continue;
      #pragma unroll
      for (int dx = -1; dx <= 1; ++dx) {
        const int xx = x + dx;
        if (xx < 0 || xx >= W_) continue;
        acc += wp[(dy + 1) * 3 + (dx + 1)] * fp[(size_t)yy * W_ + xx];
      }
    }
  }
  float* dst = which ? freq : coef;
  dst[((size_t)b * HID_ + o) * NPIX + p] = acc;
}

// ---------------------------------------------------------------------------
// 5) transpose + fp16-convert MLP weights
// ---------------------------------------------------------------------------
__global__ void k_wt(const float* __restrict__ w1, const float* __restrict__ w2,
                     _Float16* __restrict__ W1T, _Float16* __restrict__ W2T) {
  const int idx = blockIdx.x * blockDim.x + threadIdx.x;
  if (idx < 256 * HID_) {
    const int n = idx / HID_, k = idx % HID_;
    W1T[idx] = (_Float16)w1[(size_t)k * 256 + n];
  } else if (idx < 256 * HID_ + 256 * 256) {
    const int i2 = idx - 256 * HID_;
    const int n = i2 / 256, k = i2 % 256;
    W2T[i2] = (_Float16)w2[(size_t)k * 256 + n];
  }
}

// ---------------------------------------------------------------------------
// 6) fused LIIF query kernel — one wave32 per block, 64 queries (4 M-tiles).
//    B fragments held in registers across 4 WMMAs (4x L2-traffic reduction).
//    GEMM2 split into two 128-wide N halves re-using the A buffer as H2 so
//    everything fits in 64KB static LDS; final 256->3 layer fused per half.
// ---------------------------------------------------------------------------
__global__ void __launch_bounds__(32) k_liif(
    const float* __restrict__ coef, const float* __restrict__ freq,
    const float* __restrict__ inp_l, const float* __restrict__ coord,
    const float* __restrict__ cell, const float* __restrict__ w_phase,
    const _Float16* __restrict__ W1T, const _Float16* __restrict__ W2T,
    const float* __restrict__ w3, const float* __restrict__ b1,
    const float* __restrict__ b2, const float* __restrict__ b3,
    float* __restrict__ out) {
  // padded strides: 136/264 halves -> 272/528B rows (16B aligned, bank-rotated)
  __shared__ __align__(16) _Float16 Alds[64][136];  // A tile, then H2 halves
  __shared__ __align__(16) _Float16 H1[64][264];    // hidden1 (f16)
  __shared__ float CY[64], CX[64], RC0[64], RC1[64];
  __shared__ int   LIN[64];
  __shared__ float R0[64], R1[64];
  __shared__ float AREA[4][64];
  __shared__ float PRED[4][64][3];

  const int lane = threadIdx.x & 31;
  const int grp  = lane >> 4;   // half-wave group (K-split per ISA A/B layout)
  const int ml   = lane & 15;   // row-in-Mtile (A/C) and column n (B/C)

  const int tile = blockIdx.x;              // 64-query tile
  const int b    = tile / (Q_ / 64);
  const int qb   = (tile % (Q_ / 64)) * 64;

  // per-query base state (each lane serves queries m = lane, lane+32)
  #pragma unroll
  for (int mh = 0; mh < 2; ++mh) {
    const int m = mh * 32 + lane;
    const size_t qo = ((size_t)b * Q_ + (qb + m)) * 2;
    CY[m]  = coord[qo + 0];
    CX[m]  = coord[qo + 1];
    RC0[m] = cell[qo + 0] * (float)H_;
    RC1[m] = cell[qo + 1] * (float)W_;
  }
  __syncthreads();

  const size_t chanBase = (size_t)b * HID_ * NPIX;
  float pr[4][2][3];   // pred per shift per (mh) — written per shift to PRED

  for (int s = 0; s < 4; ++s) {
    const float vx = (s & 2) ? 1.0f : -1.0f;   // outer loop of reference
    const float vy = (s & 1) ? 1.0f : -1.0f;   // inner loop of reference

    // ---- per-query shift geometry ----
    #pragma unroll
    for (int mh = 0; mh < 2; ++mh) {
      const int m = mh * 32 + lane;
      const float cy = CY[m], cx = CX[m];
      float cys = cy + vx * (1.0f / H_) + 1e-6f;
      float cxs = cx + vy * (1.0f / W_) + 1e-6f;
      cys = fminf(fmaxf(cys, -1.0f + 1e-6f), 1.0f - 1e-6f);
      cxs = fminf(fmaxf(cxs, -1.0f + 1e-6f), 1.0f - 1e-6f);
      const int iy = min(max((int)roundf((cys + 1.0f) * (H_ * 0.5f) - 0.5f), 0), H_ - 1);
      const int ix = min(max((int)roundf((cxs + 1.0f) * (W_ * 0.5f) - 0.5f), 0), W_ - 1);
      const float qy = -1.0f + (2.0f * iy + 1.0f) / H_;
      const float qx = -1.0f + (2.0f * ix + 1.0f) / W_;
      const float rel0 = (cy - qy) * (float)H_;
      const float rel1 = (cx - qx) * (float)W_;
      LIN[m] = iy * W_ + ix;
      R0[m] = rel0; R1[m] = rel1;
      AREA[s][m] = fabsf(rel0 * rel1) + 1e-9f;
    }
    __syncthreads();

    // ---- fill A tile: x[m][c] = coef*(freq*posenc(rel)[c] + phase[c]) ----
    for (int c = 0; c < HID_; ++c) {          // c uniform -> scalar math/loads
      const int cm = c & 63;
      const int j  = cm & 31;
      const float fmul = exp2f((10.0f / 31.0f) * (float)j) * PI_F;
      const float wp0 = w_phase[2 * c], wp1 = w_phase[2 * c + 1];
      #pragma unroll
      for (int mh = 0; mh < 2; ++mh) {
        const int m = mh * 32 + lane;
        const int lin = LIN[m];
        const float r = (c < 64) ? R0[m] : R1[m];
        const float a = r * fmul;
        const float pe = (cm < 32) ? __sinf(a) : __cosf(a);
        const float ph = RC0[m] * wp0 + RC1[m] * wp1;
        const float cf = coef[chanBase + (size_t)c * NPIX + lin];
        const float fq = freq[chanBase + (size_t)c * NPIX + lin];
        Alds[m][c] = (_Float16)(cf * (fq * pe + ph));
      }
    }
    __syncthreads();

    // ---- GEMM1: [64x128] x [128x256] + b1, ReLU -> H1 ----
    for (int nt = 0; nt < 16; ++nt) {
      const int n = nt * 16 + ml;
      const float bias = b1[n];
      v8f acc[4];
      #pragma unroll
      for (int mt = 0; mt < 4; ++mt)
        #pragma unroll
        for (int r = 0; r < 8; ++r) acc[mt][r] = bias;
      #pragma unroll
      for (int kt = 0; kt < 4; ++kt) {
        const int kb = kt * 32 + grp * 8;
        union { v16h v; v8h h[2]; } Bf;
        Bf.h[0] = *(const v8h*)&W1T[(size_t)n * HID_ + kb];
        Bf.h[1] = *(const v8h*)&W1T[(size_t)n * HID_ + kb + 16];
        #pragma unroll
        for (int mt = 0; mt < 4; ++mt) {       // 4x B-fragment reuse
          union { v16h v; v8h h[2]; } Af;
          Af.h[0] = *(const v8h*)&Alds[mt * 16 + ml][kb];
          Af.h[1] = *(const v8h*)&Alds[mt * 16 + ml][kb + 16];
          acc[mt] = __builtin_amdgcn_wmma_f32_16x16x32_f16(
              false, Af.v, false, Bf.v, (short)0, acc[mt], false, false);
        }
      }
      #pragma unroll
      for (int mt = 0; mt < 4; ++mt)
        #pragma unroll
        for (int r = 0; r < 8; ++r)
          H1[mt * 16 + r + grp * 8][n] = (_Float16)fmaxf(acc[mt][r], 0.0f);
    }
    __syncthreads();

    // ---- GEMM2 in two 128-wide halves; H2 frags land in Alds; fused 256->3 ----
    float p00 = b3[0], p01 = b3[1], p02 = b3[2];   // pred for m = lane
    float p10 = b3[0], p11 = b3[1], p12 = b3[2];   // pred for m = lane+32
    for (int half = 0; half < 2; ++half) {
      for (int nt8 = 0; nt8 < 8; ++nt8) {
        const int n = (half * 8 + nt8) * 16 + ml;
        const float bias = b2[n];
        v8f acc[4];
        #pragma unroll
        for (int mt = 0; mt < 4; ++mt)
          #pragma unroll
          for (int r = 0; r < 8; ++r) acc[mt][r] = bias;
        #pragma unroll
        for (int kt = 0; kt < 8; ++kt) {
          const int kb = kt * 32 + grp * 8;
          union { v16h v; v8h h[2]; } Bf;
          Bf.h[0] = *(const v8h*)&W2T[(size_t)n * 256 + kb];
          Bf.h[1] = *(const v8h*)&W2T[(size_t)n * 256 + kb + 16];
          #pragma unroll
          for (int mt = 0; mt < 4; ++mt) {
            union { v16h v; v8h h[2]; } Af;
            Af.h[0] = *(const v8h*)&H1[mt * 16 + ml][kb];
            Af.h[1] = *(const v8h*)&H1[mt * 16 + ml][kb + 16];
            acc[mt] = __builtin_amdgcn_wmma_f32_16x16x32_f16(
                false, Af.v, false, Bf.v, (short)0, acc[mt], false, false);
          }
        }
        const int nloc = nt8 * 16 + ml;
        #pragma unroll
        for (int mt = 0; mt < 4; ++mt)
          #pragma unroll
          for (int r = 0; r < 8; ++r)
            Alds[mt * 16 + r + grp * 8][nloc] = (_Float16)fmaxf(acc[mt][r], 0.0f);
      }
      __syncthreads();
      // partial output-layer accumulation over this 128-wide half
      for (int n = 0; n < 128; ++n) {
        const int ng = half * 128 + n;
        const float w30 = w3[ng * 3 + 0], w31 = w3[ng * 3 + 1], w32 = w3[ng * 3 + 2];
        const float h0 = (float)Alds[lane][n];
        const float h1 = (float)Alds[lane + 32][n];
        p00 += h0 * w30; p01 += h0 * w31; p02 += h0 * w32;
        p10 += h1 * w30; p11 += h1 * w31; p12 += h1 * w32;
      }
      __syncthreads();
    }
    PRED[s][lane][0] = p00; PRED[s][lane][1] = p01; PRED[s][lane][2] = p02;
    PRED[s][lane + 32][0] = p10; PRED[s][lane + 32][1] = p11; PRED[s][lane + 32][2] = p12;
    __syncthreads();
    (void)pr;
  }

  // ---- diagonal-swapped area blend + bilinear-border skip from inp_l ----
  #pragma unroll
  for (int mh = 0; mh < 2; ++mh) {
    const int m = mh * 32 + lane;
    const int q = qb + m;
    const float a0 = AREA[0][m], a1 = AREA[1][m], a2 = AREA[2][m], a3 = AREA[3][m];
    const float inv = 1.0f / (a0 + a1 + a2 + a3);
    const float wgt[4] = { a3 * inv, a2 * inv, a1 * inv, a0 * inv };
    float r0 = 0.f, r1 = 0.f, r2 = 0.f;
    #pragma unroll
    for (int s = 0; s < 4; ++s) {
      r0 += PRED[s][m][0] * wgt[s];
      r1 += PRED[s][m][1] * wgt[s];
      r2 += PRED[s][m][2] * wgt[s];
    }
    const float cy = CY[m], cx = CX[m];
    const float fy = (cy + 1.0f) * (H_ * 0.5f) - 0.5f;
    const float fx = (cx + 1.0f) * (W_ * 0.5f) - 0.5f;
    const float y0f = floorf(fy), x0f = floorf(fx);
    const float wy = fy - y0f, wx = fx - x0f;
    const int y0 = min(max((int)y0f, 0), H_ - 1);
    const int y1 = min(max((int)y0f + 1, 0), H_ - 1);
    const int x0 = min(max((int)x0f, 0), W_ - 1);
    const int x1 = min(max((int)x0f + 1, 0), W_ - 1);
    const float* il = inp_l + (size_t)b * 3 * NPIX;
    const float rr[3] = { r0, r1, r2 };
    #pragma unroll
    for (int ch = 0; ch < 3; ++ch) {
      const float* pch = il + (size_t)ch * NPIX;
      const float v00 = pch[y0 * W_ + x0], v01 = pch[y0 * W_ + x1];
      const float v10 = pch[y1 * W_ + x0], v11 = pch[y1 * W_ + x1];
      const float top = v00 * (1.0f - wx) + v01 * wx;
      const float bot = v10 * (1.0f - wx) + v11 * wx;
      const float bil = top * (1.0f - wy) + bot * wy;
      out[((size_t)b * Q_ + q) * 3 + ch] = rr[ch] + bil;
    }
  }
}

// ---------------------------------------------------------------------------
extern "C" void kernel_launch(void* const* d_in, const int* in_sizes, int n_in,
                              void* d_out, int out_size, void* d_ws, size_t ws_size,
                              hipStream_t stream) {
  (void)in_sizes; (void)n_in; (void)out_size; (void)ws_size;
  const float* feat_left  = (const float*)d_in[0];
  const float* feat_right = (const float*)d_in[1];
  const float* M          = (const float*)d_in[2];
  const float* inp_l      = (const float*)d_in[3];
  const float* w_projr    = (const float*)d_in[4];
  const float* b_projr    = (const float*)d_in[5];
  const float* w_coef     = (const float*)d_in[6];
  const float* b_coef     = (const float*)d_in[7];
  const float* w_freq     = (const float*)d_in[8];
  const float* b_freq     = (const float*)d_in[9];
  const float* w_phase    = (const float*)d_in[10];
  const float* w1         = (const float*)d_in[11];
  const float* b1         = (const float*)d_in[12];
  const float* w2         = (const float*)d_in[13];
  const float* b2         = (const float*)d_in[14];
  const float* w3         = (const float*)d_in[15];
  const float* b3         = (const float*)d_in[16];
  const float* coord      = (const float*)d_in[17];
  const float* cell       = (const float*)d_in[18];
  float* out = (float*)d_out;

  // workspace layout (all offsets 256B aligned)
  char* ws = (char*)d_ws;
  float*    att  = (float*)(ws + 0);               // 7,077,888 B
  float*    fr   = (float*)(ws + 7077888);         // 4,718,592 B
  float*    feat = (float*)(ws + 11796480);        // 4,718,592 B
  float*    coef = (float*)(ws + 16515072);        // 9,437,184 B
  float*    freq = (float*)(ws + 25952256);        // 9,437,184 B
  _Float16* W1T  = (_Float16*)(ws + 35389440);     // 65,536 B
  _Float16* W2T  = (_Float16*)(ws + 35454976);     // 131,072 B
  // total = 35,586,048 bytes

  k_softmax<<<(B_ * H_ * W_) / 4, 128, 0, stream>>>(M, att);
  k_proj<<<(B_ * C_ * NPIX + 255) / 256, 256, 0, stream>>>(feat_right, w_projr, b_projr, fr);
  k_warp<<<(B_ * C_ * NPIX + 255) / 256, 256, 0, stream>>>(feat_left, fr, att, feat);
  k_conv3<<<(2 * B_ * HID_ * NPIX + 255) / 256, 256, 0, stream>>>(
      feat, w_coef, b_coef, w_freq, b_freq, coef, freq);
  k_wt<<<(256 * HID_ + 256 * 256 + 255) / 256, 256, 0, stream>>>(w1, w2, W1T, W2T);
  // 4608 64-query tiles, one wave32 per block
  k_liif<<<B_ * Q_ / 64, 32, 0, stream>>>(
      coef, freq, inp_l, coord, cell, w_phase, W1T, W2T, w3, b1, b2, b3, out);
}
